// _HFExaoneAttention_25245817766063
// MI455X (gfx1250) — matrix operand order repaired and checked by reading, not compiled, not run on hardware
//
#include <hip/hip_runtime.h>
#include <hip/hip_bf16.h>

typedef __bf16 bf16;
typedef __bf16 v16bf __attribute__((ext_vector_type(16)));
typedef float  v8f   __attribute__((ext_vector_type(8)));
typedef unsigned int u32x4 __attribute__((ext_vector_type(4)));
typedef int          i32x4 __attribute__((ext_vector_type(4)));
typedef int          i32x8 __attribute__((ext_vector_type(8)));

#define NHEADS 32
#define NKVH   8
#define HDIM   128
#define SEQ    2048
#define BATCH  2
#define DMODEL 4096
#define ATT_SCALE 0.08838834764831845f

#if defined(__HIP_DEVICE_COMPILE__) && __has_builtin(__builtin_amdgcn_tensor_load_to_lds)
#define USE_TDM 1
#else
#define USE_TDM 0
#endif

static __device__ __forceinline__ v8f wmma_bf16(v16bf a, v16bf b, v8f c) {
  return __builtin_amdgcn_wmma_f32_16x16x32_bf16(false, a, false, b, (short)0, c, false, false);
}

#if USE_TDM
// Issue one 2D TDM tile load: tileRows x tileElems bf16 elements from a row-major
// tensor (rowElems elements per row) into LDS at lds_off, padding the LDS dest
// 16B after every 64B (row stride 32 elems data + 8 elems pad = 40 bf16 = 80B).
static __device__ __forceinline__ void tdm_load_tile_2d(
    const bf16* gptr, unsigned lds_off, unsigned rows, unsigned rowElems,
    unsigned tileRows, unsigned tileElems)
{
  unsigned long long ga = (unsigned long long)(size_t)gptr;
  u32x4 g0;
  g0[0] = 1u;                                   // count=1 valid descriptor
  g0[1] = lds_off;                              // lds_addr (bytes)
  g0[2] = (unsigned)(ga & 0xFFFFFFFFu);         // global_addr[31:0]
  g0[3] = (unsigned)((ga >> 32) & 0x01FFFFFFu)  // global_addr[56:32]
          | (2u << 30);                         // type=2 ("image")
  i32x8 g1;
  g1[0] = (int)((1u << 16)      // data_size = 1 (2-byte elements)
              | (1u << 20)      // pad_enable
              | (3u << 22)      // pad_interval: 16 DWORDs (64B) between pads
              | (3u << 25));    // pad_amount:   4 DWORDs (16B) of pad
  g1[1] = (int)((rowElems & 0xFFFFu) << 16);                    // tensor_dim0 lo
  g1[2] = (int)((rowElems >> 16) | ((rows & 0xFFFFu) << 16));   // dim0 hi | dim1 lo
  g1[3] = (int)((rows >> 16) | (tileElems << 16));              // dim1 hi | tile_dim0
  g1[4] = (int)tileRows;                                        // tile_dim1 (tile_dim2=0)
  g1[5] = (int)rowElems;                                        // tensor_dim0_stride lo
  g1[6] = 0;
  g1[7] = 0;
  i32x4 gz = {0, 0, 0, 0};
#if __clang_major__ >= 23
  i32x8 gz8 = {0, 0, 0, 0, 0, 0, 0, 0};
  __builtin_amdgcn_tensor_load_to_lds(g0, g1, gz, gz, gz8, 0);
#else
  __builtin_amdgcn_tensor_load_to_lds(g0, g1, gz, gz, 0);
#endif
}
#endif

// ---------------- fp32 -> bf16 conversion ----------------
__global__ __launch_bounds__(256) void cvt_f32_bf16(const float* __restrict__ src,
                                                    bf16* __restrict__ dst, int n) {
  int i = (blockIdx.x * 256 + threadIdx.x) * 4;
  if (i + 3 < n) {
    float4 v = *(const float4*)(src + i);
    dst[i + 0] = (bf16)v.x;
    dst[i + 1] = (bf16)v.y;
    dst[i + 2] = (bf16)v.z;
    dst[i + 3] = (bf16)v.w;
  }
}

// ---------------- GEMM: C[M,N] = A[M,K] * Bw[N,K]^T ----------------
// 256 thr = 8 waves; block tile 128x128, K step 32; wave tile 32x64 (2x4 WMMA accs).
// A/B tiles double-buffered in LDS; on CDNA5 the Tensor Data Mover DMAs tile i+1
// while WMMAs consume tile i (TENSORcnt pipelining).
template <int WRITE_F32>
__global__ __launch_bounds__(256) void gemm_bf16_nt(
    const bf16* __restrict__ A, const bf16* __restrict__ Bw,
    bf16* __restrict__ Cb, float* __restrict__ Cf,
    int M, int N, int K)
{
  __shared__ __align__(16) bf16 As[2][128 * 40];
  __shared__ __align__(16) bf16 Bs[2][128 * 40];

  int tid  = threadIdx.x;
  int lane = tid & 31, l = lane & 15, hi = lane >> 4;
  int wave = tid >> 5;
  int wm = wave >> 1, wn = wave & 1;
  int bm = blockIdx.y * 128, bn = blockIdx.x * 128;

  v8f acc[2][4] = {};
  const int NK = K >> 5;

#if USE_TDM
  if (wave == 0) {
    tdm_load_tile_2d(A  + (size_t)bm * K, (unsigned)(size_t)&As[0][0],
                     (unsigned)M, (unsigned)K, 128u, 32u);
    tdm_load_tile_2d(Bw + (size_t)bn * K, (unsigned)(size_t)&Bs[0][0],
                     (unsigned)N, (unsigned)K, 128u, 32u);
  }
#endif

  for (int it = 0; it < NK; ++it) {
    int buf = it & 1;
#if USE_TDM
    if (wave == 0) {
      if (it + 1 < NK) {
        int k1 = (it + 1) << 5;
        tdm_load_tile_2d(A  + (size_t)bm * K + k1, (unsigned)(size_t)&As[buf ^ 1][0],
                         (unsigned)M, (unsigned)K, 128u, 32u);
        tdm_load_tile_2d(Bw + (size_t)bn * K + k1, (unsigned)(size_t)&Bs[buf ^ 1][0],
                         (unsigned)N, (unsigned)K, 128u, 32u);
        __builtin_amdgcn_s_wait_tensorcnt(2);   // current A+B pair done; next pair in flight
      } else {
        __builtin_amdgcn_s_wait_tensorcnt(0);
      }
    }
#else
    {
      int k0 = it << 5;
#pragma unroll
      for (int j = 0; j < 2; ++j) {
        int chunk = tid * 2 + j;          // 0..511
        int row = chunk >> 2;
        int cc  = (chunk & 3) << 3;       // 0,8,16,24
        *(uint4*)&As[buf][row * 40 + cc] = *(const uint4*)&A [(size_t)(bm + row) * K + k0 + cc];
        *(uint4*)&Bs[buf][row * 40 + cc] = *(const uint4*)&Bw[(size_t)(bn + row) * K + k0 + cc];
      }
    }
#endif
    __syncthreads();

    v16bf af[2], bfr[4];
#pragma unroll
    for (int t = 0; t < 2; ++t) {
      const bf16* ap = &As[buf][(wm * 32 + t * 16 + l) * 40];
#pragma unroll
      for (int i = 0; i < 8; ++i) {
        int kb = (i < 4) ? (hi * 8 + 2 * i) : (16 + hi * 8 + 2 * (i - 4));
        af[t][2 * i]     = ap[kb];
        af[t][2 * i + 1] = ap[kb + 1];
      }
    }
#pragma unroll
    for (int u = 0; u < 4; ++u) {
      const bf16* bp = &Bs[buf][(wn * 64 + u * 16 + l) * 40];
#pragma unroll
      for (int i = 0; i < 8; ++i) {
        int kb = (i < 4) ? (hi * 8 + 2 * i) : (16 + hi * 8 + 2 * (i - 4));
        bfr[u][2 * i]     = bp[kb];
        bfr[u][2 * i + 1] = bp[kb + 1];
      }
    }
#pragma unroll
    for (int t = 0; t < 2; ++t)
#pragma unroll
      for (int u = 0; u < 4; ++u)
        acc[t][u] = wmma_bf16(af[t], bfr[u], acc[t][u]);
    __syncthreads();
  }

  // straight-line epilogue (output type is a compile-time choice)
#pragma unroll
  for (int t = 0; t < 2; ++t) {
#pragma unroll
    for (int u = 0; u < 4; ++u) {
      int row0 = bm + wm * 32 + t * 16 + hi * 8;
      int col  = bn + wn * 64 + u * 16 + l;
#pragma unroll
      for (int r = 0; r < 8; ++r) {
        float v = acc[t][u][r];
        if (WRITE_F32) Cf[(size_t)(row0 + r) * N + col] = v;
        else           Cb[(size_t)(row0 + r) * N + col] = (bf16)v;
      }
    }
  }
}

// ---------------- RoPE + layout pack ----------------
__global__ __launch_bounds__(256) void rope_pack(
    const bf16* __restrict__ Qg, const bf16* __restrict__ Kg, const bf16* __restrict__ Vg,
    const float* __restrict__ cosp, const float* __restrict__ sinp,
    bf16* __restrict__ Qr, bf16* __restrict__ Kr, bf16* __restrict__ Vr)
{
  int gid  = blockIdx.x * 8 + (threadIdx.x >> 5);
  int lane = threadIdx.x & 31;
  int hslot = gid % 48;
  int bs    = gid / 48;
  int b = bs / SEQ, s = bs % SEQ;
  int d = lane * 2;

  const bf16* src;
  bf16* dst;
  bool dorope = true;
  if (hslot < NHEADS) {
    src = Qg + (size_t)bs * (NHEADS * HDIM) + hslot * HDIM;
    dst = Qr + (((size_t)b * NHEADS + hslot) * SEQ + s) * HDIM;
  } else if (hslot < NHEADS + NKVH) {
    int h = hslot - NHEADS;
    src = Kg + (size_t)bs * (NKVH * HDIM) + h * HDIM;
    dst = Kr + (((size_t)b * NKVH + h) * SEQ + s) * HDIM;
  } else {
    int h = hslot - NHEADS - NKVH;
    src = Vg + (size_t)bs * (NKVH * HDIM) + h * HDIM;
    dst = Vr + (((size_t)b * NKVH + h) * SEQ + s) * HDIM;
    dorope = false;
  }
#pragma unroll
  for (int j = 0; j < 2; ++j) {
    int dd = d + j;
    float x0 = (float)src[dd];
    float x1 = (float)src[dd + 64];
    if (dorope) {
      float c  = cosp[(size_t)bs * HDIM + dd];
      float sn = sinp[(size_t)bs * HDIM + dd];
      dst[dd]      = (bf16)(x0 * c - x1 * sn);
      dst[dd + 64] = (bf16)(x1 * c + x0 * sn);
    } else {
      dst[dd]      = (bf16)x0;
      dst[dd + 64] = (bf16)x1;
    }
  }
}

// ---------------- causal flash attention ----------------
__global__ __launch_bounds__(128) void flash_attn(
    const bf16* __restrict__ Qr, const bf16* __restrict__ Kr, const bf16* __restrict__ Vr,
    bf16* __restrict__ AO)
{
  __shared__ __align__(16) float sS[4 * 16 * 33];

  int tid  = threadIdx.x;
  int wave = tid >> 5, lane = tid & 31, l = lane & 15, hi = lane >> 4;
  int bh = blockIdx.y, b = bh >> 5, h = bh & 31, kvh = h >> 2;   // GROUPS=4
  int q0b = blockIdx.x * 64;
  int q0  = q0b + wave * 16;

  const bf16* Qp = Qr + (((size_t)b * NHEADS + h)   * SEQ + q0) * HDIM;
  const bf16* Kp = Kr + (((size_t)b * NKVH  + kvh) * SEQ)      * HDIM;
  const bf16* Vp = Vr + (((size_t)b * NKVH  + kvh) * SEQ)      * HDIM;
  float* sW = &sS[wave * 16 * 33];

  v16bf qf[4];
#pragma unroll
  for (int c = 0; c < 4; ++c)
#pragma unroll
    for (int i = 0; i < 8; ++i) {
      int kb = (i < 4) ? (hi * 8 + 2 * i) : (16 + hi * 8 + 2 * (i - 4));
      qf[c][2 * i]     = Qp[(size_t)l * HDIM + c * 32 + kb];
      qf[c][2 * i + 1] = Qp[(size_t)l * HDIM + c * 32 + kb + 1];
    }

  float m_r[8], l_r[8];
  v8f oacc[8] = {};
#pragma unroll
  for (int r = 0; r < 8; ++r) { m_r[r] = -3.0e38f; l_r[r] = 0.0f; }

  int kend = q0b + 64;   // same trip count for all waves (barrier-safe)
  for (int kc = 0; kc < kend; kc += 32) {
    v8f sacc[2] = {};
#pragma unroll
    for (int t = 0; t < 2; ++t)
#pragma unroll
      for (int c = 0; c < 4; ++c) {
        v16bf kf;
        const bf16* kp = Kp + (size_t)(kc + t * 16 + l) * HDIM + c * 32;
#pragma unroll
        for (int i = 0; i < 8; ++i) {
          int kk = (i < 4) ? (hi * 8 + 2 * i) : (16 + hi * 8 + 2 * (i - 4));
          kf[2 * i]     = kp[kk];
          kf[2 * i + 1] = kp[kk + 1];
        }
        sacc[t] = wmma_bf16(qf[c], kf, sacc[t]);
      }

#pragma unroll
    for (int r = 0; r < 8; ++r) {
      int qrow = q0 + hi * 8 + r;
      float s0 = sacc[0][r] * ATT_SCALE;
      float s1 = sacc[1][r] * ATT_SCALE;
      if (kc + l > qrow)      s0 = -3.0e38f;
      if (kc + 16 + l > qrow) s1 = -3.0e38f;
      float mx = fmaxf(s0, s1);
#pragma unroll
      for (int off = 8; off >= 1; off >>= 1)
        mx = fmaxf(mx, __shfl_xor(mx, off, 16));
      float mn    = fmaxf(m_r[r], mx);
      float alpha = __expf(m_r[r] - mn);
      float p0    = __expf(s0 - mn);
      float p1    = __expf(s1 - mn);
      float rs    = p0 + p1;
#pragma unroll
      for (int off = 8; off >= 1; off >>= 1)
        rs += __shfl_xor(rs, off, 16);
      l_r[r] = l_r[r] * alpha + rs;
      m_r[r] = mn;
#pragma unroll
      for (int o = 0; o < 8; ++o) oacc[o][r] = oacc[o][r] * alpha;
      sW[(hi * 8 + r) * 33 + l]      = p0;
      sW[(hi * 8 + r) * 33 + 16 + l] = p1;
    }
    __syncthreads();

    v16bf pf;
#pragma unroll
    for (int i = 0; i < 8; ++i) {
      int kb = (i < 4) ? (hi * 8 + 2 * i) : (16 + hi * 8 + 2 * (i - 4));
      pf[2 * i]     = (bf16)sW[l * 33 + kb];
      pf[2 * i + 1] = (bf16)sW[l * 33 + kb + 1];
    }
#pragma unroll
    for (int o = 0; o < 8; ++o) {
      v16bf vf;
#pragma unroll
      for (int i = 0; i < 8; ++i) {
        int kk = (i < 4) ? (hi * 8 + 2 * i) : (16 + hi * 8 + 2 * (i - 4));
        vf[2 * i]     = Vp[(size_t)(kc + kk)     * HDIM + o * 16 + l];
        vf[2 * i + 1] = Vp[(size_t)(kc + kk + 1) * HDIM + o * 16 + l];
      }
      oacc[o] = wmma_bf16(pf, vf, oacc[o]);
    }
    __syncthreads();
  }

#pragma unroll
  for (int o = 0; o < 8; ++o) {
#pragma unroll
    for (int r = 0; r < 8; ++r) {
      int row = q0 + hi * 8 + r;
      float inv = 1.0f / l_r[r];
      AO[((size_t)b * SEQ + row) * (NHEADS * HDIM) + h * HDIM + o * 16 + l] =
          (bf16)(oacc[o][r] * inv);
    }
  }
}

// ---------------- launch ----------------
extern "C" void kernel_launch(void* const* d_in, const int* in_sizes, int n_in,
                              void* d_out, int out_size, void* d_ws, size_t ws_size,
                              hipStream_t stream) {
  (void)in_sizes; (void)n_in; (void)out_size; (void)ws_size;
  const float* hs   = (const float*)d_in[0];
  const float* cosp = (const float*)d_in[1];
  const float* sinp = (const float*)d_in[2];
  const float* wq   = (const float*)d_in[3];
  const float* wk   = (const float*)d_in[4];
  const float* wv   = (const float*)d_in[5];
  const float* wo   = (const float*)d_in[6];
  float* out = (float*)d_out;

  const size_t MR = (size_t)BATCH * SEQ;   // 4096 rows
  const size_t K  = DMODEL;                // 4096

  char* ws = (char*)d_ws;
  size_t off = 0;
  auto alloc = [&](size_t elems) {
    bf16* p = (bf16*)(ws + off);
    off += ((elems * 2 + 255) & ~(size_t)255);
    return p;
  };
  bf16* Xb  = alloc(MR * K);
  bf16* Wqb = alloc((size_t)(NHEADS * HDIM) * K);
  bf16* Wkb = alloc((size_t)(NKVH * HDIM) * K);
  bf16* Wvb = alloc((size_t)(NKVH * HDIM) * K);
  bf16* Wob = alloc((size_t)DMODEL * (NHEADS * HDIM));
  bf16* Qg  = alloc(MR * NHEADS * HDIM);
  bf16* Kg  = alloc(MR * NKVH * HDIM);
  bf16* Vg  = alloc(MR * NKVH * HDIM);
  bf16* Qr  = alloc(MR * NHEADS * HDIM);
  bf16* Kr  = alloc(MR * NKVH * HDIM);
  bf16* Vr  = alloc(MR * NKVH * HDIM);
  bf16* AO  = alloc(MR * NHEADS * HDIM);

  auto cvt = [&](const float* s, bf16* d, size_t n) {
    cvt_f32_bf16<<<dim3((unsigned)(n / 1024)), dim3(256), 0, stream>>>(s, d, (int)n);
  };
  cvt(hs, Xb, MR * K);
  cvt(wq, Wqb, (size_t)(NHEADS * HDIM) * K);
  cvt(wk, Wkb, (size_t)(NKVH * HDIM) * K);
  cvt(wv, Wvb, (size_t)(NKVH * HDIM) * K);
  cvt(wo, Wob, (size_t)DMODEL * (NHEADS * HDIM));

  gemm_bf16_nt<0><<<dim3(NHEADS * HDIM / 128, (unsigned)(MR / 128)), 256, 0, stream>>>(
      Xb, Wqb, Qg, nullptr, (int)MR, NHEADS * HDIM, (int)K);
  gemm_bf16_nt<0><<<dim3(NKVH * HDIM / 128, (unsigned)(MR / 128)), 256, 0, stream>>>(
      Xb, Wkb, Kg, nullptr, (int)MR, NKVH * HDIM, (int)K);
  gemm_bf16_nt<0><<<dim3(NKVH * HDIM / 128, (unsigned)(MR / 128)), 256, 0, stream>>>(
      Xb, Wvb, Vg, nullptr, (int)MR, NKVH * HDIM, (int)K);

  rope_pack<<<dim3((unsigned)(MR * 48 / 8)), 256, 0, stream>>>(
      Qg, Kg, Vg, cosp, sinp, Qr, Kr, Vr);

  flash_attn<<<dim3(SEQ / 64, BATCH * NHEADS), 128, 0, stream>>>(Qr, Kr, Vr, AO);

  gemm_bf16_nt<1><<<dim3(DMODEL / 128, (unsigned)(MR / 128)), 256, 0, stream>>>(
      AO, Wob, nullptr, out, (int)MR, DMODEL, NHEADS * HDIM);
}
